// convVAE_7078106104495
// MI455X (gfx1250) — matrix operand order repaired and checked by reading, not compiled, not run on hardware
//
#include <hip/hip_runtime.h>
#include <math.h>

// ---------------- problem constants (from reference) ----------------
#define NUM_GRAPHS 2048
#define NODES      64
#define NTOT       (NUM_GRAPHS * NODES)   // 131072
#define ETOT       (NTOT * 8)             // 1048576
#define EPG        (ETOT / NUM_GRAPHS)    // 512 edges per graph
#define IN_DIM     64
#define H1D        128
#define H2D        64
#define OUT_DIM    2016

// frag-packed f16 weight offsets in workspace (units: halves)
#define OFF_W1REL  0
#define OFF_W1ROOT 8192
#define OFF_W2REL  16384
#define OFF_W2ROOT 24576
#define OFF_W3REL  32768
#define OFF_W3ROOT 40960
#define OFF_FC1    49152
#define OFF_FC2    57344
#define WF_TOTAL   315392      // + fc2 (128*2016)
#define OFF_Z      WF_TOTAL    // z: [2048][64] f16 row-major

typedef __attribute__((ext_vector_type(16))) _Float16 v16h;
typedef __attribute__((ext_vector_type(8)))  float    v8f;

union Frag16 { v16h h; unsigned int u[8]; uint4 q[2]; _Float16 e[16]; };
union Acc8   { v8f f; float e[8]; };

#define WMMA_F16(A, B, C) \
  __builtin_amdgcn_wmma_f32_16x16x32_f16(false, (A), false, (B), (short)0, (C), false, false)

// ---- A-operand fragment (16x32 f16) from row-major [rows][stride] source ----
// lanes 0-15 -> M=lane, VGPR v holds K=2v,2v+1 (K=0..15);
// lanes 16-31: same M, K=16..31. Pairs are contiguous -> u32 loads.
__device__ __forceinline__ v16h load_afrag(const _Float16* base, int stride,
                                           int row0, int k0, int lane) {
  const int sub = (lane >> 4) & 1;
  const _Float16* p = base + (row0 + (lane & 15)) * stride + k0;
  Frag16 f;
#pragma unroll
  for (int v = 0; v < 8; ++v) {
    int kb = 16 * sub + 2 * v;
    f.u[v] = *(const unsigned int*)(p + kb);
  }
  return f.h;
}

// ---- B-operand fragment (32x16 f16) from row-major TRANSPOSED source Bt[N][K] ----
// lanes 0-15: column n=lane, K=0..15 (2 per VGPR); lanes 16-31: same columns, K=16..31.
__device__ __forceinline__ v16h load_bfrag(const _Float16* bt, int stride,
                                           int n0, int k0, int lane) {
  const int sub = (lane >> 4) & 1;
  const _Float16* p = bt + (n0 + (lane & 15)) * stride + k0 + sub * 16;
  Frag16 f;
#pragma unroll
  for (int v = 0; v < 8; ++v) f.u[v] = *(const unsigned int*)(p + 2 * v);
  return f.h;
}

// ---- B fragment from frag-packed global weights: lane's 16 halves contiguous ----
__device__ __forceinline__ v16h load_wfrag(const _Float16* wf, int off, int blk, int lane) {
  const _Float16* p = wf + off + (blk * 32 + lane) * 16;
  Frag16 f;
  f.q[0] = *(const uint4*)p;
  f.q[1] = *(const uint4*)(p + 8);
  return f.h;
}

// ================= kernel 1: repack fp32 weights -> f16 B-fragment order =========
__global__ __launch_bounds__(256) void vae_prep_weights(
    const float* w1_rel, const float* w1_root, const float* w2_rel, const float* w2_root,
    const float* w3_rel, const float* w3_root, const float* fc1_w, const float* fc2_w,
    _Float16* wf) {
  int idx = blockIdx.x * blockDim.x + threadIdx.x;
  if (idx >= WF_TOTAL) return;
  const int   sizes[8] = {8192, 8192, 8192, 8192, 8192, 8192, 8192, 258048};
  const int   Ks[8]    = {64, 64, 128, 128, 128, 128, 64, 128};
  const int   Ns[8]    = {128, 128, 64, 64, 64, 64, 128, 2016};
  const float* srcs[8] = {w1_rel, w1_root, w2_rel, w2_root, w3_rel, w3_root, fc1_w, fc2_w};
  int m = 0, off = 0;
  while (idx - off >= sizes[m]) { off += sizes[m]; ++m; }
  int li = idx - off;
  int blk = li >> 9, lane = (li >> 4) & 31, h = li & 15;
  int KC = Ks[m] >> 5;
  int nt = blk / KC, kc = blk - nt * KC;
  int n = nt * 16 + (lane & 15);
  int k = kc * 32 + ((lane >> 4) & 1) * 16 + h;   // B layout: lane-half selects K 0-15/16-31
  wf[idx] = (_Float16)srcs[m][k * Ns[m] + n];
}

// ================= kernel 2: fused per-graph encoder (1 block = 1 graph) =========
__global__ __launch_bounds__(128) void vae_encoder(
    const float* __restrict__ x, const int* __restrict__ ei, const float* __restrict__ eps,
    const float* __restrict__ b1, const float* __restrict__ b2, const float* __restrict__ b3,
    const _Float16* __restrict__ wf, _Float16* __restrict__ zf,
    float* __restrict__ mu_out, float* __restrict__ lv_out) {
  __shared__ __align__(16) unsigned char smem[57856];
  _Float16* sA    = (_Float16*)(smem);          // 8 KB: adjacency counts -> f16
  _Float16* sX    = (_Float16*)(smem + 8192);   // 8 KB
  _Float16* sXt   = (_Float16*)(smem + 16384);  // 8 KB, reused as AggX
  _Float16* sAggH = (_Float16*)(smem + 8192);   // 16 KB, reuses X+Xt (dead after phase 2)
  _Float16* sH1   = (_Float16*)(smem + 24576);  // 16 KB
  _Float16* sH1t  = (_Float16*)(smem + 40960);  // 16 KB
  float*    sMu   = (float*)(smem + 57344);     // 256 B
  float*    sLv   = (float*)(smem + 57600);     // 256 B

  const int tid  = threadIdx.x;
  const int g    = blockIdx.x;
  const int lane = tid & 31;
  const int wave = tid >> 5;
  const int ln15 = lane & 15;
  const int sub  = (lane >> 4) & 1;
  const int row0 = wave * 16;

  // ---- phase 0: zero counters, stage X (and X^T) as f16 ----
  unsigned int* Acnt = (unsigned int*)sA;   // 64x32 dwords: two u16 counters per dword
  for (int w = tid; w < 2048; w += 128) Acnt[w] = 0u;
  if (tid < 64) { sMu[tid] = 0.f; sLv[tid] = 0.f; }
  for (int idx = tid; idx < NODES * IN_DIM; idx += 128) {
    int r = idx >> 6, c = idx & 63;
    _Float16 h = (_Float16)x[(g * NODES + r) * IN_DIM + c];
    sX[r * 64 + c] = h;
    sXt[c * 64 + r] = h;
  }
  __syncthreads();

  // ---- build adjacency counts: A[i][j] += 1 for edge j->i (dst aggregates src) ----
  const int e0 = g * EPG;
  for (int t = tid; t < EPG; t += 128) {
    int js = ei[e0 + t] & 63;          // src local
    int is = ei[ETOT + e0 + t] & 63;   // dst local
    atomicAdd(&Acnt[is * 32 + (js >> 1)], (js & 1) ? 0x10000u : 1u);
  }
  __syncthreads();
  // convert packed counts -> packed f16 in place (max count 512, exact in f16)
  for (int w = tid; w < 2048; w += 128) {
    unsigned int c = Acnt[w];
    Frag16 pk;
    pk.e[0] = (_Float16)(float)(c & 0xffffu);
    pk.e[1] = (_Float16)(float)(c >> 16);
    Acnt[w] = pk.u[0];
  }
  __syncthreads();

  // ---- phase 1: AggX = A @ X  (M=16 band per wave, K=64, N=64) ----
  Frag16 fA0, fA1;
  fA0.h = load_afrag(sA, 64, row0, 0, lane);
  fA1.h = load_afrag(sA, 64, row0, 32, lane);
  Acc8 aggx[4];
#pragma unroll
  for (int nt = 0; nt < 4; ++nt) {
    Acc8 a; a.f = (v8f)0.f;
    a.f = WMMA_F16(fA0.h, load_bfrag(sXt, 64, nt * 16, 0, lane), a.f);
    a.f = WMMA_F16(fA1.h, load_bfrag(sXt, 64, nt * 16, 32, lane), a.f);
    aggx[nt] = a;
  }
  __syncthreads();                    // everyone done reading sXt
#pragma unroll
  for (int nt = 0; nt < 4; ++nt)
#pragma unroll
    for (int v = 0; v < 8; ++v)
      sXt[(row0 + v + 8 * sub) * 64 + nt * 16 + ln15] = (_Float16)aggx[nt].e[v]; // sXt = AggX now
  __syncthreads();

  // ---- phase 2: H1 = relu(AggX@W1rel + X@W1root + b1)  (N=128, K=64) ----
  Frag16 fAgg[2], fX[2];
#pragma unroll
  for (int kc = 0; kc < 2; ++kc) {
    fAgg[kc].h = load_afrag(sXt, 64, row0, kc * 32, lane);
    fX[kc].h   = load_afrag(sX, 64, row0, kc * 32, lane);
  }
#pragma unroll
  for (int nt = 0; nt < 8; ++nt) {
    Acc8 a; a.f = (v8f)0.f;
#pragma unroll
    for (int kc = 0; kc < 2; ++kc) {
      a.f = WMMA_F16(fAgg[kc].h, load_wfrag(wf, OFF_W1REL, nt * 2 + kc, lane), a.f);
      a.f = WMMA_F16(fX[kc].h,   load_wfrag(wf, OFF_W1ROOT, nt * 2 + kc, lane), a.f);
    }
    float bn = b1[nt * 16 + ln15];
#pragma unroll
    for (int v = 0; v < 8; ++v) {
      float hv = a.e[v] + bn;
      hv = hv > 0.f ? hv : 0.f;
      _Float16 hh = (_Float16)hv;
      int m = row0 + v + 8 * sub, n = nt * 16 + ln15;
      sH1[m * 128 + n] = hh;
      sH1t[n * 64 + m] = hh;
    }
  }
  __syncthreads();                    // H1/H1t complete; X/AggX now dead

  // ---- phase 3: AggH1 = A @ H1  (N=128, K=64), store into region over X/Xt ----
#pragma unroll
  for (int nt = 0; nt < 8; ++nt) {
    Acc8 a; a.f = (v8f)0.f;
    a.f = WMMA_F16(fA0.h, load_bfrag(sH1t, 64, nt * 16, 0, lane), a.f);
    a.f = WMMA_F16(fA1.h, load_bfrag(sH1t, 64, nt * 16, 32, lane), a.f);
#pragma unroll
    for (int v = 0; v < 8; ++v)
      sAggH[(row0 + v + 8 * sub) * 128 + nt * 16 + ln15] = (_Float16)a.e[v];
  }
  __syncthreads();

  // ---- phase 4: MuPre/LvPre = AggH1@W{2,3}rel + H1@W{2,3}root + b  -> column mean ----
  Frag16 fAH[4], fH[4];
#pragma unroll
  for (int kc = 0; kc < 4; ++kc) {
    fAH[kc].h = load_afrag(sAggH, 128, row0, kc * 32, lane);
    fH[kc].h  = load_afrag(sH1, 128, row0, kc * 32, lane);
  }
#pragma unroll
  for (int nt = 0; nt < 4; ++nt) {
    Acc8 aM, aL; aM.f = (v8f)0.f; aL.f = (v8f)0.f;
#pragma unroll
    for (int kc = 0; kc < 4; ++kc) {
      aM.f = WMMA_F16(fAH[kc].h, load_wfrag(wf, OFF_W2REL, nt * 4 + kc, lane), aM.f);
      aM.f = WMMA_F16(fH[kc].h,  load_wfrag(wf, OFF_W2ROOT, nt * 4 + kc, lane), aM.f);
      aL.f = WMMA_F16(fAH[kc].h, load_wfrag(wf, OFF_W3REL, nt * 4 + kc, lane), aL.f);
      aL.f = WMMA_F16(fH[kc].h,  load_wfrag(wf, OFF_W3ROOT, nt * 4 + kc, lane), aL.f);
    }
    float sM = 8.f * b2[nt * 16 + ln15];   // 8 rows per lane each carry the bias
    float sL = 8.f * b3[nt * 16 + ln15];
#pragma unroll
    for (int v = 0; v < 8; ++v) { sM += aM.e[v]; sL += aL.e[v]; }
    sM += __shfl_xor(sM, 16, 32);          // combine the two 8-row halves of this band
    sL += __shfl_xor(sL, 16, 32);
    if (lane < 16) {
      atomicAdd(&sMu[nt * 16 + lane], sM);
      atomicAdd(&sLv[nt * 16 + lane], sL);
    }
  }
  __syncthreads();

  // ---- finalize: mean, reparameterize, emit ----
  if (tid < 64) {
    float mu = sMu[tid] * (1.f / 64.f);
    float lv = sLv[tid] * (1.f / 64.f);
    int gi = g * 64 + tid;
    mu_out[gi] = mu;
    lv_out[gi] = lv;
    float z = mu + eps[gi] * expf(0.5f * lv);
    zf[gi] = (_Float16)z;
  }
}

// ================= kernel 3: decoder  recon = sigmoid(relu(z@fc1+b)@fc2+b) ======
__global__ __launch_bounds__(128) void vae_decoder(
    const _Float16* __restrict__ zf, const _Float16* __restrict__ wf,
    const float* __restrict__ fc1_b, const float* __restrict__ fc2_b,
    float* __restrict__ recon) {
  __shared__ __align__(16) _Float16 sH[4][16 * 128];   // per-wave h band, 16 KB
  const int tid = threadIdx.x, lane = tid & 31, wave = tid >> 5;
  const int ln15 = lane & 15, sub = (lane >> 4) & 1;
  const int row0 = blockIdx.x * 64 + wave * 16;        // global z row base

  // z A-fragments straight from global f16 row-major [2048][64]
  Frag16 fZ[2];
#pragma unroll
  for (int kc = 0; kc < 2; ++kc) fZ[kc].h = load_afrag(zf, 64, row0, kc * 32, lane);

  // h = relu(z @ fc1 + fc1_b)   (M=16, K=64, N=128)
  _Float16* myH = sH[wave];
#pragma unroll
  for (int nt = 0; nt < 8; ++nt) {
    Acc8 a; a.f = (v8f)0.f;
#pragma unroll
    for (int kc = 0; kc < 2; ++kc)
      a.f = WMMA_F16(fZ[kc].h, load_wfrag(wf, OFF_FC1, nt * 2 + kc, lane), a.f);
    float bn = fc1_b[nt * 16 + ln15];
#pragma unroll
    for (int v = 0; v < 8; ++v) {
      float hv = a.e[v] + bn;
      myH[(v + 8 * sub) * 128 + nt * 16 + ln15] = (_Float16)(hv > 0.f ? hv : 0.f);
    }
  }
  // wave-local LDS RAW: compiler inserts s_wait_dscnt; reload as A-fragments (K=128)
  Frag16 fHh[4];
#pragma unroll
  for (int kc = 0; kc < 4; ++kc) fHh[kc].h = load_afrag(myH, 128, 0, kc * 32, lane);

  // recon tile loop: 126 n-tiles of 16 over OUT_DIM=2016, K=128 (4 chunks)
  for (int nt = 0; nt < OUT_DIM / 16; ++nt) {
    Acc8 a; a.f = (v8f)0.f;
#pragma unroll
    for (int kc = 0; kc < 4; ++kc)
      a.f = WMMA_F16(fHh[kc].h, load_wfrag(wf, OFF_FC2, nt * 4 + kc, lane), a.f);
    int n = nt * 16 + ln15;
    float bn = fc2_b[n];
#pragma unroll
    for (int v = 0; v < 8; ++v) {
      float val = a.e[v] + bn;
      recon[(row0 + v + 8 * sub) * OUT_DIM + n] = 1.f / (1.f + expf(-val));
    }
  }
}

// ================================ launch =========================================
extern "C" void kernel_launch(void* const* d_in, const int* in_sizes, int n_in,
                              void* d_out, int out_size, void* d_ws, size_t ws_size,
                              hipStream_t stream) {
  const float* x       = (const float*)d_in[0];
  const int*   ei      = (const int*)d_in[1];
  // d_in[2] = batch (unused: graphs are contiguous blocks of 64 nodes)
  const float* eps     = (const float*)d_in[3];
  const float* w1_rel  = (const float*)d_in[4];
  const float* b1      = (const float*)d_in[5];
  const float* w1_root = (const float*)d_in[6];
  const float* w2_rel  = (const float*)d_in[7];
  const float* b2      = (const float*)d_in[8];
  const float* w2_root = (const float*)d_in[9];
  const float* w3_rel  = (const float*)d_in[10];
  const float* b3      = (const float*)d_in[11];
  const float* w3_root = (const float*)d_in[12];
  const float* fc1_w   = (const float*)d_in[13];
  const float* fc1_b   = (const float*)d_in[14];
  const float* fc2_w   = (const float*)d_in[15];
  const float* fc2_b   = (const float*)d_in[16];

  _Float16* wf = (_Float16*)d_ws;
  _Float16* zf = wf + OFF_Z;

  float* out   = (float*)d_out;
  float* recon = out;                               // [2048][2016]
  float* mu    = out + NUM_GRAPHS * OUT_DIM;        // [2048][64]
  float* lv    = mu + NUM_GRAPHS * H2D;             // [2048][64]

  vae_prep_weights<<<WF_TOTAL / 256, 256, 0, stream>>>(
      w1_rel, w1_root, w2_rel, w2_root, w3_rel, w3_root, fc1_w, fc2_w, wf);
  vae_encoder<<<NUM_GRAPHS, 128, 0, stream>>>(x, ei, eps, b1, b2, b3, wf, zf, mu, lv);
  vae_decoder<<<NUM_GRAPHS / 64, 128, 0, stream>>>(zf, wf, fc1_b, fc2_b, recon);
}
